// GradientBasedOptimizer_76982993814144
// MI455X (gfx1250) — compile-verified, matrix operand order checked by reading, not run on hardware
//
#include <hip/hip_runtime.h>
#include <hip/hip_bf16.h>

// ---------------------------------------------------------------------------
// Problem constants (from the reference)
// ---------------------------------------------------------------------------
#define BATCH   2
#define SEQ     4096          // T == S
#define DM      1024          // D_MODEL
#define HEADS   2
#define DIM     32            // head dim == WMMA K for bf16 (16x16x32)
#define ROWS    (BATCH * SEQ) // 8192 flattened (b,t) rows
#define NQK     96            // 64 q cols + 32 k cols packed together

typedef __attribute__((ext_vector_type(16))) __bf16 v16bf;
typedef __attribute__((ext_vector_type(8)))  float  v8f;

union FragB16 {          // one 16x32 (or 32x16) bf16 WMMA operand: 8 VGPRs
    v16bf v;
    uint4 q[2];          // two 16-byte chunks per lane
};

__device__ __forceinline__ unsigned short f2bf(float f) {
    unsigned int u = __float_as_uint(f);
    unsigned int r = u + 0x7FFFu + ((u >> 16) & 1u);   // round to nearest even
    return (unsigned short)(r >> 16);
}

// ---------------------------------------------------------------------------
// 1a. x (f32) -> x_bf16
// ---------------------------------------------------------------------------
__global__ void convert_x_kernel(const float* __restrict__ x,
                                 unsigned short* __restrict__ xb, int n) {
    int i = blockIdx.x * blockDim.x + threadIdx.x;
    if (i < n) xb[i] = f2bf(x[i]);
}

// ---------------------------------------------------------------------------
// 1b. pack [Wq(64x1024); Wk(32x1024)] -> Wqk_bf16 [96][1024]
// ---------------------------------------------------------------------------
__global__ void convert_w_kernel(const float* __restrict__ Wq,
                                 const float* __restrict__ Wk,
                                 unsigned short* __restrict__ wb) {
    int i = blockIdx.x * blockDim.x + threadIdx.x;
    if (i >= NQK * DM) return;
    int r = i / DM, c = i - r * DM;
    float v = (r < 64) ? Wq[r * DM + c] : Wk[(r - 64) * DM + c];
    wb[i] = f2bf(v);
}

// ---------------------------------------------------------------------------
// 2. qk = x @ [Wq;Wk]^T  via bf16 WMMA.  One wave per 16x16 output tile.
//    A lane layout (16-bit A 16x32): lanes 0-15 hold M rows, lane-half selects
//    K chunks {0..7,16..23} vs {8..15,24..31}; B mirrors with N on lanes.
// ---------------------------------------------------------------------------
__global__ void proj_qk_kernel(const unsigned short* __restrict__ xb,
                               const unsigned short* __restrict__ wb,
                               unsigned short* __restrict__ qk) {
    const int tile    = blockIdx.x;          // 512 row-tiles * 6 col-tiles
    const int rowTile = tile / 6;
    const int colTile = tile - rowTile * 6;
    const int lane = threadIdx.x & 31;
    const int m    = lane & 15;
    const int half = lane >> 4;

    const unsigned short* aRow = xb + (size_t)(rowTile * 16 + m) * DM;
    const unsigned short* bRow = wb + (size_t)(colTile * 16 + m) * DM;

    v8f acc = {};
    for (int k0 = 0; k0 < DM; k0 += 32) {
        FragB16 a, b;
        a.q[0] = *(const uint4*)(aRow + k0 +      half * 8);
        a.q[1] = *(const uint4*)(aRow + k0 + 16 + half * 8);
        b.q[0] = *(const uint4*)(bRow + k0 +      half * 8);
        b.q[1] = *(const uint4*)(bRow + k0 + 16 + half * 8);
        acc = __builtin_amdgcn_wmma_f32_16x16x32_bf16(
                  false, a.v, false, b.v, (short)0, acc, false, false);
    }
    // C layout: VGPR j -> row M = j + half*8 ; col N = lane&15
#pragma unroll
    for (int j = 0; j < 8; ++j) {
        int r = rowTile * 16 + half * 8 + j;
        int c = colTile * 16 + m;
        qk[(size_t)r * NQK + c] = f2bf(acc[j]);
    }
}

// ---------------------------------------------------------------------------
// 3. wproj = x @ Ww^T  (8192x2 outputs; negligible FLOPs -> plain VALU)
// ---------------------------------------------------------------------------
__global__ void wproj_kernel(const float* __restrict__ x,
                             const float* __restrict__ Ww,
                             float* __restrict__ wp) {
    int i = blockIdx.x * blockDim.x + threadIdx.x;   // 0 .. ROWS*HEADS-1
    if (i >= ROWS * HEADS) return;
    int row = i >> 1, h = i & 1;
    const float* xr = x  + (size_t)row * DM;
    const float* wr = Ww + (size_t)h   * DM;
    float s = 0.f;
    for (int k = 0; k < DM; ++k) s += xr[k] * wr[k];
    wp[i] = s;
}

// ---------------------------------------------------------------------------
// 4. dots + relu + head-weighted sum.
//    Block = 256 threads = 8 waves; block tile = 128(t) x 128(s) of out[b].
//    The 128x32 bf16 k-tile (8 KB) is staged ONCE per block into LDS with
//    CDNA5 async copies (GLOBAL_LOAD_ASYNC_TO_LDS_B128, ASYNCcnt-tracked).
//    The LDS address operand is derived FROM the shared array's pointer so
//    the array escapes into the asm (whose "memory" clobber then counts as a
//    potential writer) and the ds_load_b128 reads are not folded away.
// ---------------------------------------------------------------------------
__global__ void dots_kernel(const unsigned short* __restrict__ qk,
                            const float* __restrict__ wp,
                            const float* __restrict__ aw,
                            float* __restrict__ out) {
    // k-tile in LDS: [128 rows][32 cols] bf16, row-major (row stride 64 B).
    __shared__ unsigned short ktile[128 * 32];

    // Insurance against "never-stored global" load folding: blockDim.x is
    // 256 at every launch, so this store never executes, but the compiler
    // cannot prove blockDim.x != 257 (any value 1..1024 is possible).
    if (blockDim.x == 257) ktile[0] = 1;

    const int b    = blockIdx.z;
    const int t0   = blockIdx.x * 128;
    const int s0   = blockIdx.y * 128;
    const int tid  = threadIdx.x;
    const int wave = tid >> 5;
    const int lane = tid & 31;
    const int m    = lane & 15;
    const int half = lane >> 4;
    const int tb   = t0 + wave * 16;

    // ---- async-stage the k tile: 512 x 16B chunks, 2 per thread -----------
#pragma unroll
    for (int c = 0; c < 2; ++c) {
        int chunk = tid + c * 256;               // 0..511
        int row   = chunk >> 2;                  // 0..127
        int ci    = chunk & 3;                   // 16B chunk within row
        const unsigned short* gp =
            qk + (size_t)(b * SEQ + s0 + row) * NQK + 64 + ci * 8;
        unsigned long long ga = (unsigned long long)(uintptr_t)gp;
        // Low 32 bits of a generic LDS pointer are the LDS byte address
        //; this also escapes `ktile`.
        unsigned int la = (unsigned int)(uintptr_t)(&ktile[chunk * 8]);
        asm volatile("global_load_async_to_lds_b128 %0, %1, off"
                     :: "v"(la), "v"(ga)
                     : "memory");
    }
    // hint the next s-block's k rows toward L2 while we compute this one
    __builtin_prefetch((const void*)(qk + (size_t)(b * SEQ + s0 + 128 + (tid >> 1)) * NQK + 64), 0, 0);

    // ---- A fragments for head 0 (cols 0..31) and head 1 (cols 32..63) -----
    const unsigned short* qrow = qk + (size_t)(b * SEQ + tb + m) * NQK;
    FragB16 a0, a1;
    a0.q[0] = *(const uint4*)(qrow +  0 + half * 8);
    a0.q[1] = *(const uint4*)(qrow + 16 + half * 8);
    a1.q[0] = *(const uint4*)(qrow + 32 + half * 8);
    a1.q[1] = *(const uint4*)(qrow + 48 + half * 8);

    // per-row head weights (row M = half*8 + j), pre-scaled by adaptation wt
    const float aw0 = aw[0], aw1 = aw[1];
    float w0[8], w1[8];
#pragma unroll
    for (int j = 0; j < 8; ++j) {
        int r = b * SEQ + tb + half * 8 + j;
        w0[j] = wp[r * 2 + 0] * aw0;
        w1[j] = wp[r * 2 + 1] * aw1;
    }

    // ---- wait for this wave's async copies, then block-wide barrier -------
    asm volatile("s_wait_asynccnt 0x0" ::: "memory");
    __syncthreads();

    // ---- 8 s-subtiles: B fragment from LDS, 2 WMMAs, relu+combine, store --
#pragma unroll
    for (int ss = 0; ss < 8; ++ss) {
        FragB16 bf;                       // B = k^T, shared by both heads
        const unsigned short* lrow = &ktile[(ss * 16 + m) * 32];
        bf.q[0] = *(const uint4*)(lrow +      half * 8);
        bf.q[1] = *(const uint4*)(lrow + 16 + half * 8);

        v8f c0 = {}, c1 = {};
        c0 = __builtin_amdgcn_wmma_f32_16x16x32_bf16(
                 false, a0.v, false, bf.v, (short)0, c0, false, false);
        c1 = __builtin_amdgcn_wmma_f32_16x16x32_bf16(
                 false, a1.v, false, bf.v, (short)0, c1, false, false);

        float* orow = out + (size_t)(b * SEQ + tb + half * 8) * SEQ
                          + (s0 + ss * 16 + m);
#pragma unroll
        for (int j = 0; j < 8; ++j) {
            float v = fmaxf(c0[j], 0.f) * w0[j] + fmaxf(c1[j], 0.f) * w1[j];
            orow[(size_t)j * SEQ] = v;
        }
    }
}

// ---------------------------------------------------------------------------
// launch
// ---------------------------------------------------------------------------
extern "C" void kernel_launch(void* const* d_in, const int* in_sizes, int n_in,
                              void* d_out, int out_size, void* d_ws, size_t ws_size,
                              hipStream_t stream) {
    const float* x  = (const float*)d_in[0];   // [2,4096,1024]
    const float* Wq = (const float*)d_in[1];   // [64,1024]
    const float* Wk = (const float*)d_in[2];   // [32,1024]
    const float* Ww = (const float*)d_in[3];   // [2,1024]
    const float* aw = (const float*)d_in[4];   // [2]
    float* out = (float*)d_out;                // [2,4096,4096]

    // workspace carve-up (all 16B aligned)
    char* ws = (char*)d_ws;
    unsigned short* xb = (unsigned short*)(ws);                         // 16 MB
    unsigned short* wb = (unsigned short*)(ws + (size_t)ROWS * DM * 2); // 192 KB
    unsigned short* qk = (unsigned short*)(ws + (size_t)ROWS * DM * 2
                                              + (size_t)NQK  * DM * 2); // 1.5 MB
    float*          wp = (float*)(ws + (size_t)ROWS * DM * 2
                                     + (size_t)NQK  * DM * 2
                                     + (size_t)ROWS * NQK * 2);         // 64 KB

    {   // 1a: x -> bf16
        int n = ROWS * DM;
        convert_x_kernel<<<(n + 255) / 256, 256, 0, stream>>>(x, xb, n);
    }
    {   // 1b: pack weights -> bf16
        int n = NQK * DM;
        convert_w_kernel<<<(n + 255) / 256, 256, 0, stream>>>(Wq, Wk, wb);
    }
    {   // 2: qk projection GEMM (one wave per 16x16 tile)
        int tiles = (ROWS / 16) * (NQK / 16);   // 512 * 6 = 3072
        proj_qk_kernel<<<tiles, 32, 0, stream>>>(xb, wb, qk);
    }
    {   // 3: w projection
        int n = ROWS * HEADS;
        wproj_kernel<<<(n + 255) / 256, 256, 0, stream>>>(x, Ww, wp);
    }
    {   // 4: dots + relu + weighted head sum
        dim3 grid(SEQ / 128, SEQ / 128, BATCH);   // (32, 32, 2)
        dots_kernel<<<grid, 256, 0, stream>>>(qk, wp, aw, out);
    }
}
